// SelfAttentionLayer_69002944577540
// MI455X (gfx1250) — matrix operand order repaired and checked by reading, not compiled, not run on hardware
//
#include <hip/hip_runtime.h>
#include <stdint.h>

// ---------------- problem constants ----------------
#define EMB    128
#define HID    256
#define NHEAD  4
#define NG     36      // number of groups
#define BB     4
#define SS     512
#define NSEQ   (BB*NG) // 144
#define DIN    384     // 3*EMB
#define G4     (4*HID) // 1024 gate width

// ---------------- types ----------------
typedef __attribute__((ext_vector_type(16))) __bf16 v16bf;
typedef __attribute__((ext_vector_type(8)))  float  v8f;
typedef __attribute__((ext_vector_type(4)))  unsigned int u32x4;
typedef __attribute__((ext_vector_type(4)))  float  f32x4;
typedef int v4i_ __attribute__((vector_size(16)));   // matches async-LDS builtin param type

union Frag { v16bf v; u32x4 q[2]; };

__device__ __forceinline__ unsigned short f2bf(float f) {
  unsigned int u = __float_as_uint(f);
  u += 0x7fffu + ((u >> 16) & 1u);          // round-to-nearest-even
  return (unsigned short)(u >> 16);
}
__device__ __forceinline__ float bf2f(unsigned short h) {
  return __uint_as_float(((unsigned int)h) << 16);
}
__device__ __forceinline__ float sigm(float x) { return 1.f / (1.f + __expf(-x)); }
__device__ __forceinline__ float tanh_f(float x) {
  float e = __expf(2.f * x);
  return (e - 1.f) / (e + 1.f);
}

// 16-byte async global->LDS copy (ASYNCcnt-tracked, bypasses VGPRs)
__device__ __forceinline__ void async_copy16(const void* g, void* l) {
#if defined(__has_builtin) && __has_builtin(__builtin_amdgcn_global_load_async_to_lds_b128)
  __builtin_amdgcn_global_load_async_to_lds_b128(
      (__attribute__((address_space(1))) v4i_*)(uintptr_t)g,
      (__attribute__((address_space(3))) v4i_*)(unsigned)(uintptr_t)l,
      0, 0);
#else
  unsigned lds = (unsigned)(uintptr_t)l;   // generic->LDS: low 32 bits are the LDS offset
  asm volatile("global_load_async_to_lds_b128 %0, %1, off" :: "v"(lds), "v"(g) : "memory");
#endif
}
__device__ __forceinline__ void wait_async0() {
  asm volatile("s_wait_asynccnt 0x0" ::: "memory");
}

// ---------------- tiny prep kernels ----------------
__global__ void k_cvt(const float* __restrict__ x, unsigned short* __restrict__ y, int n) {
  int i = blockIdx.x * 256 + threadIdx.x;
  if (i < n) y[i] = f2bf(x[i]);
}

// pack a 128-wide column slice of W_ih (1024x384) into contiguous bf16 (1024x128)
__global__ void k_pack_wih(const float* __restrict__ Wih, unsigned short* __restrict__ out, int colOff) {
  int i = blockIdx.x * 256 + threadIdx.x;
  if (i < G4 * EMB) {
    int n = i / EMB, k = i % EMB;
    out[i] = f2bf(Wih[(size_t)n * DIN + colOff + k]);
  }
}

// Cg[g][n] = rel_table[min(g,31)] . W_ih[n, 256:384] + b_ih[n] + b_hh[n]
__global__ void k_ctx(const float* __restrict__ rel, const float* __restrict__ Wih,
                      const float* __restrict__ bih, const float* __restrict__ bhh,
                      float* __restrict__ Cg) {
  int i = blockIdx.x * 256 + threadIdx.x;
  if (i >= NG * G4) return;
  int g = i / G4, n = i % G4;
  int ri = (g < 31) ? g : 31;
  const float* r = rel + (size_t)ri * EMB;
  const float* w = Wih + (size_t)n * DIN + 2 * EMB;
  float s = bih[n] + bhh[n];
  for (int k = 0; k < EMB; ++k) s += r[k] * w[k];
  Cg[i] = s;
}

// Wcat[o, h*256+k] = sum_e Wf[o, e*4+h] * Wp[e*4+h, k]   (fold Wf o Wp per head)
__global__ void k_wcat(const float* __restrict__ Wf, const float* __restrict__ Wp,
                       unsigned short* __restrict__ Wcat) {
  int i = blockIdx.x * 256 + threadIdx.x;
  if (i >= EMB * G4) return;
  int o = i / G4; int j = i % G4; int h = j / HID; int k = j % HID;
  float s = 0.f;
  for (int e = 0; e < EMB; ++e)
    s += Wf[(size_t)o * (NHEAD * EMB) + e * NHEAD + h] * Wp[((size_t)(e * NHEAD + h)) * HID + k];
  Wcat[i] = f2bf(s);
}

// outbias[o] = bf[o] + sum_j Wf[o,j]*bp[j]   (softmax weights sum to 1 over groups)
__global__ void k_outbias(const float* __restrict__ Wf, const float* __restrict__ bp,
                          const float* __restrict__ bfb, float* __restrict__ ob) {
  int o = threadIdx.x;
  if (o < EMB) {
    float s = bfb[o];
    for (int j = 0; j < NHEAD * EMB; ++j) s += Wf[(size_t)o * (NHEAD * EMB) + j] * bp[j];
    ob[o] = s;
  }
}

__global__ void k_addb(const float* __restrict__ a, const float* __restrict__ b,
                       float* __restrict__ c, int n) {
  int i = blockIdx.x * 256 + threadIdx.x;
  if (i < n) c[i] = a[i] + b[i];
}

// ---------------- generic WMMA GEMM: C = act(A(MxK) * W(NxK)^T + bias [+ extra]) ----------------
// extra_mode 1: addend = Tb[b*S + (s-d_g)] gathered per row (zero if s<d_g); row m -> (b,g,s)
__global__ __launch_bounds__(256) void gemm_wmma(
    const unsigned short* __restrict__ A, const unsigned short* __restrict__ W,
    const float* __restrict__ bias, const float* __restrict__ extra,
    float* __restrict__ outF, unsigned short* __restrict__ outH,
    int M, int N, int K, int relu, int extra_mode)
{
  int wave = threadIdx.x >> 5;
  int lane = threadIdx.x & 31;
  int tile = blockIdx.x * 8 + wave;
  int tilesN = N >> 4;
  int tm = tile / tilesN, tn = tile % tilesN;
  if (tm >= (M >> 4)) return;
  int hi = lane >> 4;
  int m = (tm << 4) + (lane & 15);
  int n = (tn << 4) + (lane & 15);
  const u32x4* pa = (const u32x4*)(A + (size_t)m * K);
  const u32x4* pb = (const u32x4*)(W + (size_t)n * K);
  v8f acc = {0.f, 0.f, 0.f, 0.f, 0.f, 0.f, 0.f, 0.f};
  int ksteps = K >> 5;
  for (int ks = 0; ks < ksteps; ++ks) {
    int q = ks << 2;   // u32x4 index: 8 bf16 per u32x4
    Frag a, b;
    a.q[0] = pa[q + hi];
    a.q[1] = pa[q + 2 + hi];
    b.q[0] = pb[q + hi];
    b.q[1] = pb[q + 2 + hi];
    acc = __builtin_amdgcn_wmma_f32_16x16x32_bf16(false, a.v, false, b.v,
                                                  (short)0, acc, false, false);
  }
  float bn = bias ? bias[n] : 0.f;
  int m0 = (tm << 4) + (hi << 3);
#pragma unroll
  for (int r = 0; r < 8; ++r) {
    int mm = m0 + r;
    float v = acc[r] + bn;
    if (extra_mode == 1) {
      int nseq = mm >> 9;           // / S
      int s = mm & (SS - 1);
      int g = nseq % NG;
      int bidx = nseq / NG;
      int d = (g < 4) ? (1 << g) : ((g - 3) << 4);
      if (s >= d) v += extra[((size_t)(bidx * SS + s - d)) * N + n];
    }
    if (relu) v = fmaxf(v, 0.f);
    size_t off = (size_t)mm * N + n;
    if (outF) outF[off] = v;
    else      outH[off] = f2bf(v);
  }
}

// ---------------- LSTM recurrence: 9 blocks x 16 sequences ----------------
// H lives in LDS (A-operand of the per-step WMMA GEMM), W_hh streamed from L2,
// gathered Tref rows for step s+1 prefetched via double-buffered ASYNC global->LDS
// copies overlapped with step s's WMMA + cell update.
__global__ __launch_bounds__(256) void lstm_kernel(
    const unsigned short* __restrict__ Whh,   // (1024,256) bf16
    const float* __restrict__ Tref,           // (B,S,1024) f32
    const float* __restrict__ Ttgt,           // (B,S,1024) f32
    const float* __restrict__ Cg,             // (36,1024) f32
    unsigned short* __restrict__ hs)          // (144,512,256) bf16, shifted by one
{
  __shared__ float gates[16][G4];             // 64 KB
  __shared__ float xbuf[2][16][G4];           // 128 KB async landing (Tref rows)
  __shared__ unsigned short Hld[16][HID];     // 8 KB
  __shared__ int sb[16], sg[16], sd[16];

  int tid = threadIdx.x;
  int wave = tid >> 5, lane = tid & 31, hi = lane >> 4;
  int seq0 = blockIdx.x * 16;
  int myseq = tid >> 4;
  int j0 = (tid & 15) << 4;

  float c[16];
#pragma unroll
  for (int r = 0; r < 16; ++r) c[r] = 0.f;

  for (int r = 0; r < 16; ++r) Hld[myseq][j0 + r] = 0;
  {  // hs[:, 0, :] = 0
    unsigned short* dst = hs + ((size_t)(seq0 + myseq) * SS) * HID + j0;
    for (int r = 0; r < 16; ++r) dst[r] = 0;
  }
  if (tid < 16) {
    int nglb = seq0 + tid;
    int b = nglb / NG, g = nglb % NG;
    sb[tid] = b; sg[tid] = g;
    sd[tid] = (g < 4) ? (1 << g) : ((g - 3) << 4);
  }
  __syncthreads();

  // issue async Tref batch for step sp into xbuf[sp&1]; 16 x b128 per thread
  auto issue_batch = [&](int sp) {
    if (sp >= SS) return;
    int sq = tid >> 4;
    int d = sd[sq];
    if (sp >= d) {
      int b = sb[sq];
      const char* src = (const char*)(Tref + ((size_t)b * SS + (sp - d)) * G4) + ((tid & 15) << 4);
      char* dst = (char*)(&xbuf[sp & 1][sq][0]) + ((tid & 15) << 4);
#pragma unroll
      for (int i = 0; i < 16; ++i)
        async_copy16(src + (i << 8), dst + (i << 8));   // row = 16 lanes x 16 chunks x 16B
    }
  };

  issue_batch(0);   // prologue prefetch

  for (int s = 0; s < SS; ++s) {
    // wait for batch(s) (everything older is done; ASYNC loads complete in order)
    wait_async0();
    __syncthreads();   // all waves' copies landed; also orders prev-iter LDS writes

    // (1) gates <- xg = xbuf[s] (Tref gather) + Ttgt[b,s] + Cg[g]
    {
      int sq = tid >> 4;
      int c0 = (tid & 15) << 6;                    // 64 floats per thread
      int b = sb[sq], g = sg[sq], d = sd[sq];
      const f32x4* t4 = (const f32x4*)(Ttgt + ((size_t)b * SS + s) * G4 + c0);
      const f32x4* c4 = (const f32x4*)(Cg + (size_t)g * G4 + c0);
      const f32x4* r4 = (const f32x4*)(&xbuf[s & 1][sq][c0]);
      f32x4* g4p = (f32x4*)(&gates[sq][c0]);
      if (s >= d) {
#pragma unroll 4
        for (int k = 0; k < 16; ++k) g4p[k] = t4[k] + c4[k] + r4[k];
      } else {
#pragma unroll 4
        for (int k = 0; k < 16; ++k) g4p[k] = t4[k] + c4[k];
      }
    }
    __syncthreads();

    // prefetch next step's Tref rows; overlaps with the WMMA burst below
    issue_batch(s + 1);

    // (2) gates += H(16x256,bf16 in LDS) * Whh^T  via WMMA; 8 N-tiles per wave
    for (int i = 0; i < 8; ++i) {
      int nt = (wave << 3) + i;
      int n = (nt << 4) + (lane & 15);
      const u32x4* pb = (const u32x4*)(Whh + (size_t)n * HID);
      const u32x4* pa = (const u32x4*)(&Hld[lane & 15][0]);
      __builtin_prefetch(pb, 0, 1);                 // global_prefetch_b8 (L2-resident Whh)
      v8f acc = {0.f, 0.f, 0.f, 0.f, 0.f, 0.f, 0.f, 0.f};
#pragma unroll
      for (int ks = 0; ks < 8; ++ks) {
        int q = ks << 2;
        Frag a, b;
        a.q[0] = pa[q + hi];
        a.q[1] = pa[q + 2 + hi];
        b.q[0] = pb[q + hi];
        b.q[1] = pb[q + 2 + hi];
        acc = __builtin_amdgcn_wmma_f32_16x16x32_bf16(false, a.v, false, b.v,
                                                      (short)0, acc, false, false);
      }
      int m0 = hi << 3;
#pragma unroll
      for (int r = 0; r < 8; ++r) gates[m0 + r][n] += acc[r];
    }
    __syncthreads();

    // (3) elementwise LSTM cell update; thread owns (myseq, j0..j0+15); c in registers
    {
      float hval[16];
#pragma unroll
      for (int r = 0; r < 16; ++r) {
        int j = j0 + r;
        float ig = gates[myseq][j];
        float fg = gates[myseq][j + 256];
        float gg = gates[myseq][j + 512];
        float og = gates[myseq][j + 768];
        float cc = sigm(fg) * c[r] + sigm(ig) * tanh_f(gg);
        c[r] = cc;
        hval[r] = sigm(og) * tanh_f(cc);
      }
      for (int r = 0; r < 16; ++r) Hld[myseq][j0 + r] = f2bf(hval[r]);
      if (s + 1 < SS) {
        unsigned short* dst = hs + (((size_t)(seq0 + myseq)) * SS + (s + 1)) * HID + j0;
        for (int r = 0; r < 16; ++r) dst[r] = f2bf(hval[r]);
      }
    }
    // barrier for Hld/gates hazards is at the top of the next iteration
  }
}

// ---------------- softmax over groups + z = sum_g attn*h2 (per head) ----------------
__global__ __launch_bounds__(256) void combine_kernel(
    const unsigned short* __restrict__ h2,   // (144,512,256) bf16
    const float* __restrict__ Ww, const float* __restrict__ bw,
    unsigned short* __restrict__ z,          // (2048, 1024) bf16 : [bs][h*256+k]
    float* __restrict__ attn_out)            // (B,G,S,1,NHEAD) f32
{
  __shared__ unsigned short sh[NG][HID];     // 18 KB
  __shared__ float wlog[NG][NHEAD];
  __shared__ float sattn[NG][NHEAD];
  int tid = threadIdx.x;
  int bs = blockIdx.x;                       // 0..2047
  int b = bs >> 9, s = bs & (SS - 1);

  for (int g = 0; g < NG; ++g) {
    size_t row = (((size_t)(b * NG + g)) * SS + s) * HID;
    sh[g][tid] = h2[row + tid];
  }
  __syncthreads();

  if (tid < NG * NHEAD) {                    // 144 logits, one dot each
    int g = tid >> 2, h = tid & 3;
    const float* w = Ww + (size_t)h * HID;
    float acc = bw[h];
    for (int k = 0; k < HID; ++k) acc += bf2f(sh[g][k]) * w[k];
    wlog[g][h] = acc;
  }
  __syncthreads();

  if (tid < NHEAD) {                         // softmax over 36 groups, per head
    int h = tid;
    float mx = -1e30f;
    for (int g = 0; g < NG; ++g) mx = fmaxf(mx, wlog[g][h]);
    float e[NG]; float sum = 0.f;
    for (int g = 0; g < NG; ++g) { e[g] = __expf(wlog[g][h] - mx); sum += e[g]; }
    float inv = 1.f / sum;
    for (int g = 0; g < NG; ++g) sattn[g][h] = e[g] * inv;
  }
  __syncthreads();

  if (tid < NG * NHEAD) {
    int g = tid >> 2, h = tid & 3;
    attn_out[(((size_t)(b * NG + g)) * SS + s) * NHEAD + h] = sattn[g][h];
  }

  float zacc[NHEAD] = {0.f, 0.f, 0.f, 0.f};
  for (int g = 0; g < NG; ++g) {
    float hv = bf2f(sh[g][tid]);
#pragma unroll
    for (int h = 0; h < NHEAD; ++h) zacc[h] += sattn[g][h] * hv;
  }
  size_t zrow = (size_t)bs * G4;
#pragma unroll
  for (int h = 0; h < NHEAD; ++h) z[zrow + h * HID + tid] = f2bf(zacc[h]);
}

// ---------------- host launcher ----------------
extern "C" void kernel_launch(void* const* d_in, const int* in_sizes, int n_in,
                              void* d_out, int out_size, void* d_ws, size_t ws_size,
                              hipStream_t stream) {
  const float* emb = (const float*)d_in[0];
  const float* rel = (const float*)d_in[1];
  const float* Wih = (const float*)d_in[2];
  const float* Whh = (const float*)d_in[3];
  const float* bih = (const float*)d_in[4];
  const float* bhh = (const float*)d_in[5];
  const float* W1a = (const float*)d_in[6];
  const float* b1a = (const float*)d_in[7];
  const float* W1b = (const float*)d_in[8];
  const float* b1b = (const float*)d_in[9];
  const float* W2  = (const float*)d_in[10];
  const float* b2  = (const float*)d_in[11];
  const float* Ww  = (const float*)d_in[12];
  const float* bw  = (const float*)d_in[13];
  const float* Wp  = (const float*)d_in[14];
  const float* bp  = (const float*)d_in[15];
  const float* Wf  = (const float*)d_in[16];
  const float* bfb = (const float*)d_in[17];

  float* out  = (float*)d_out;                    // (B,S,EMB)
  float* attn = out + (size_t)BB * SS * EMB;      // (B,G,S,1,NHEAD)

  char* w = (char*)d_ws;
  auto alloc = [&](size_t bytes) -> char* {
    char* p = w; w += (bytes + 255) & ~(size_t)255; return p;
  };
  unsigned short* embh    = (unsigned short*)alloc((size_t)BB*SS*EMB*2);       // 512 KB
  unsigned short* WihRefH = (unsigned short*)alloc((size_t)G4*EMB*2);          // 256 KB
  unsigned short* WihTgtH = (unsigned short*)alloc((size_t)G4*EMB*2);
  unsigned short* WhhH    = (unsigned short*)alloc((size_t)G4*HID*2);          // 512 KB
  unsigned short* W1aH    = (unsigned short*)alloc((size_t)HID*HID*2);
  unsigned short* W1bH    = (unsigned short*)alloc((size_t)HID*EMB*2);
  unsigned short* W2H     = (unsigned short*)alloc((size_t)HID*HID*2);
  unsigned short* WcatH   = (unsigned short*)alloc((size_t)EMB*G4*2);
  float* outbias = (float*)alloc(EMB*4);
  float* b1ab    = (float*)alloc(HID*4);
  float* Cg      = (float*)alloc((size_t)NG*G4*4);                              // 144 KB
  float* Tref    = (float*)alloc((size_t)BB*SS*G4*4);                           // 8 MB
  float* Ttgt    = (float*)alloc((size_t)BB*SS*G4*4);                           // 8 MB
  float* Tb      = (float*)alloc((size_t)BB*SS*HID*4);                          // 2 MB
  unsigned short* hsH = (unsigned short*)alloc((size_t)NSEQ*SS*HID*2);          // 37.75 MB
  unsigned short* h1H = (unsigned short*)alloc((size_t)NSEQ*SS*HID*2);          // 37.75 MB
  unsigned short* h2H = hsH;   // alias: hs is dead after h1 GEMM
  unsigned short* zH  = (unsigned short*)alloc((size_t)BB*SS*G4*2);             // 4 MB

  // --- prep / conversions ---
  k_cvt<<<(BB*SS*EMB + 255)/256, 256, 0, stream>>>(emb, embh, BB*SS*EMB);
  k_pack_wih<<<(G4*EMB + 255)/256, 256, 0, stream>>>(Wih, WihRefH, 0);
  k_pack_wih<<<(G4*EMB + 255)/256, 256, 0, stream>>>(Wih, WihTgtH, EMB);
  k_cvt<<<(G4*HID + 255)/256, 256, 0, stream>>>(Whh, WhhH, G4*HID);
  k_cvt<<<(HID*HID + 255)/256, 256, 0, stream>>>(W1a, W1aH, HID*HID);
  k_cvt<<<(HID*EMB + 255)/256, 256, 0, stream>>>(W1b, W1bH, HID*EMB);
  k_cvt<<<(HID*HID + 255)/256, 256, 0, stream>>>(W2, W2H, HID*HID);
  k_ctx<<<(NG*G4 + 255)/256, 256, 0, stream>>>(rel, Wih, bih, bhh, Cg);
  k_wcat<<<(EMB*G4 + 255)/256, 256, 0, stream>>>(Wf, Wp, WcatH);
  k_outbias<<<1, 128, 0, stream>>>(Wf, bp, bfb, outbias);
  k_addb<<<1, 256, 0, stream>>>(b1a, b1b, b1ab, HID);

  // --- input-gate precompute (structure-exploiting) ---
  // Tref = emb @ Wih[:, :128]^T ; Ttgt = emb @ Wih[:,128:256]^T   (2048x128 x 1024x128)
  gemm_wmma<<<(128*64)/8, 256, 0, stream>>>(embh, WihRefH, nullptr, nullptr,
      Tref, nullptr, BB*SS, G4, EMB, 0, 0);
  gemm_wmma<<<(128*64)/8, 256, 0, stream>>>(embh, WihTgtH, nullptr, nullptr,
      Ttgt, nullptr, BB*SS, G4, EMB, 0, 0);
  // Tb = emb @ W1b^T (bias folded into b1ab later)
  gemm_wmma<<<(128*16)/8, 256, 0, stream>>>(embh, W1bH, nullptr, nullptr,
      Tb, nullptr, BB*SS, HID, EMB, 0, 0);

  // --- LSTM recurrence (serial over 512 steps; 9 WGPs x 16 sequences) ---
  lstm_kernel<<<NSEQ/16, 256, 0, stream>>>(WhhH, Tref, Ttgt, Cg, hsH);

  // --- MLP: h1 = relu(hs@W1a^T + (b1a+b1b) + gather(Tb)) ; h2 = relu(h1@W2^T + b2) ---
  gemm_wmma<<<((NSEQ*SS/16)*(HID/16))/8, 256, 0, stream>>>(hsH, W1aH, b1ab, Tb,
      nullptr, h1H, NSEQ*SS, HID, HID, 1, 1);
  gemm_wmma<<<((NSEQ*SS/16)*(HID/16))/8, 256, 0, stream>>>(h1H, W2H, b2, nullptr,
      nullptr, h2H, NSEQ*SS, HID, HID, 1, 0);

  // --- softmax over groups + per-head z (pred tensor algebraically eliminated) ---
  combine_kernel<<<BB*SS, 256, 0, stream>>>(h2H, Ww, bw, zH, attn);

  // --- out = z_cat @ Wcat^T + outbias ---
  gemm_wmma<<<((BB*SS/16)*(EMB/16))/8, 256, 0, stream>>>(zH, WcatH, outbias, nullptr,
      out, nullptr, BB*SS, EMB, G4, 0, 0);
}
